// Net_15762529976343
// MI455X (gfx1250) — matrix operand order repaired and checked by reading, not compiled, not run on hardware
//
#include <hip/hip_runtime.h>
#include <hip/hip_bf16.h>

typedef __bf16 bf16;
typedef __attribute__((ext_vector_type(16))) __bf16 v16bf;
typedef __attribute__((ext_vector_type(8)))  float  v8f;

#if defined(__has_builtin)
#if __has_builtin(__builtin_amdgcn_global_load_async_to_lds_b128) && \
    __has_builtin(__builtin_amdgcn_s_wait_asynccnt)
#define USE_ASYNC_LDS 1
#endif
#endif

#if USE_ASYNC_LDS
// builtin signature (from hipcc diagnostic): (int4 __device__*, int4 __shared__*, Ii, Ii)
typedef int v4i_b128 __attribute__((vector_size(16)));
typedef __attribute__((address_space(1))) v4i_b128* gptr_b128;
typedef __attribute__((address_space(3))) v4i_b128* lptr_b128;
#endif

#define K_TOT 18432   // 2048 * 3 * 3
#define M_TOT 200     // B(8) * 5 * 5
#define M_PAD 208     // 13 * 16
#define CIN   2048
#define LDSW  72      // LDS row stride in bf16 (144B = 9*16B: aligned + conflict-free)
#define KSTEP 64      // two 16x16x32 WMMAs per stage

// ---------------- fp32 -> bf16 weight conversion (zero-pads rows) ----------------
__global__ __launch_bounds__(256) void cvt_bf16_kernel(
    const float* __restrict__ src, bf16* __restrict__ dst,
    unsigned long long n, unsigned long long npad) {
  unsigned long long i = (unsigned long long)blockIdx.x * blockDim.x + threadIdx.x;
  unsigned long long stride = (unsigned long long)gridDim.x * blockDim.x;
  for (; i < npad; i += stride) {
    float v = (i < n) ? src[i] : 0.f;
    dst[i] = (bf16)v;
  }
}

// ---------------- im2col pack: fp32 NCHW -> bf16 [M_PAD, K_TOT] ----------------
// One shot per GEMM stage; removes all gather math from the GEMM inner loop.
__global__ __launch_bounds__(256) void im2col_bf16_kernel(
    const float* __restrict__ A, bf16* __restrict__ Apack,
    int Hin, int Win, int stride, int pad) {
  const int k = blockIdx.x * 256 + threadIdx.x;   // grid.x = 72 -> k in [0,18432)
  const int m = blockIdx.y;                       // grid.y = 208
  float v = 0.f;
  if (m < M_TOT) {
    const int c  = k / 9;
    const int r  = k - 9 * c;
    const int kh = r / 3;
    const int kw = r - 3 * kh;
    const int b   = m / 25;
    const int pos = m - 25 * b;
    const int oh  = pos / 5;
    const int ow  = pos - 5 * oh;
    const int ih  = oh * stride - pad + kh;
    const int iw  = ow * stride - pad + kw;
    if (ih >= 0 && ih < Hin && iw >= 0 && iw < Win)
      v = A[((b * CIN + c) * Hin + ih) * Win + iw];
  }
  Apack[(size_t)m * K_TOT + k] = (bf16)v;
}

// ---------------- GEMM: Out[m,n] = sum_k Apack[m,k] * W[n,k] ----------------
// Block: 256 thr = 8 waves; tile 16(M) x 128(N); double-buffered async LDS staging.
// W must be padded to gridDim.y*128 rows. Apack padded to 208 rows.
__global__ __launch_bounds__(256) void gemm_wmma(
    const bf16* __restrict__ Apack, const bf16* __restrict__ W,
    float* __restrict__ Out, const float* __restrict__ bias,
    int Nreal, int outMode /*0: NCHW, 1: [M,N]+relu*/) {
  __shared__ bf16 As[2][16 * LDSW];
  __shared__ bf16 Bs[2][128 * LDSW];
  const int tid  = threadIdx.x;
  const int lane = tid & 31;
  const int wv   = tid >> 5;
  const int m0   = blockIdx.x << 4;
  const int n0   = blockIdx.y << 7;
  const int mrow = lane & 15;
  const int hi   = lane >> 4;      // 0/1
  const int ksel = hi << 3;        // A: K-half base 0/8
  const int koff = hi << 4;        // unused; B loads also use ksel (0/8)
  const int nloc = (wv << 4) + (lane & 15);
  const int NK   = K_TOT / KSTEP;  // 288
  const bf16* aBase = Apack + (size_t)m0 * K_TOT;
  const bf16* bBase = W + (size_t)n0 * K_TOT;
  v8f acc = {};

  auto compute = [&](int buf) {
    const bf16* as = &As[buf][0];
    const bf16* bs = &Bs[buf][0];
#pragma unroll
    for (int half = 0; half < 2; ++half) {
      const int kb = half << 5;    // 0, 32
      v16bf af, bfr;
#pragma unroll
      for (int t = 0; t < 8; ++t) {
        af[t]     = as[mrow * LDSW + kb + ksel + t];       // K = kb+ksel+t
        af[8 + t] = as[mrow * LDSW + kb + 16 + ksel + t];  // K = kb+16+ksel+t
      }
#pragma unroll
      for (int t = 0; t < 8; ++t) {
        bfr[t]     = bs[nloc * LDSW + kb + ksel + t];       // K = kb+ksel+t
        bfr[8 + t] = bs[nloc * LDSW + kb + 16 + ksel + t];  // K = kb+16+ksel+t
      }
      acc = __builtin_amdgcn_wmma_f32_16x16x32_bf16(
          false, af, false, bfr, (short)0, acc, false, false);
    }
  };

#if USE_ASYNC_LDS
  // Each wave issues exactly 5 async b128 instructions per tile (1 A + 4 B),
  // so s_wait_asynccnt 5 == "previous tile complete" (async loads finish in order).
  auto issue = [&](int it) {
    const int k0 = it * KSTEP;
    bf16* as = &As[it & 1][0];
    bf16* bs = &Bs[it & 1][0];
    {
      // 128 A chunks (16 rows x 8x16B); threads 128-255 duplicate 0-127
      // (identical bytes to identical LDS address: race-benign, no EXEC branch).
      const int cid = tid & 127;
      const int r = cid >> 3, ch = cid & 7;
      __builtin_amdgcn_global_load_async_to_lds_b128(
          (gptr_b128)(aBase + (size_t)r * K_TOT + k0 + (ch << 3)),
          (lptr_b128)(as + r * LDSW + (ch << 3)), 0, 0);
    }
#pragma unroll
    for (int q = 0; q < 4; ++q) {                // 1024 B chunks (128 rows x 8x16B)
      const int e = tid + (q << 8);
      const int r = e >> 3, ch = e & 7;
      __builtin_amdgcn_global_load_async_to_lds_b128(
          (gptr_b128)(bBase + (size_t)r * K_TOT + k0 + (ch << 3)),
          (lptr_b128)(bs + r * LDSW + (ch << 3)), 0, 0);
    }
  };

  issue(0);
  for (int it = 0; it < NK; ++it) {
    if (it + 1 < NK) {
      issue(it + 1);                              // prefetch next tile into other buffer
      __builtin_amdgcn_s_wait_asynccnt(5);        // tile `it` complete (in-order)
    } else {
      __builtin_amdgcn_s_wait_asynccnt(0);
    }
    __syncthreads();
    compute(it & 1);
    __syncthreads();
  }
#else
  for (int it = 0; it < NK; ++it) {
    const int k0 = it * KSTEP;
    for (int e = tid; e < 1152; e += 256) {
      int cid = e;
      const bf16* src; bf16* dst;
      if (cid < 128) {
        const int r = cid >> 3, ch = cid & 7;
        src = aBase + (size_t)r * K_TOT + k0 + (ch << 3);
        dst = &As[0][0] + r * LDSW + (ch << 3);
      } else {
        cid -= 128;
        const int r = cid >> 3, ch = cid & 7;
        src = bBase + (size_t)r * K_TOT + k0 + (ch << 3);
        dst = &Bs[0][0] + r * LDSW + (ch << 3);
      }
      *(uint4*)dst = *(const uint4*)src;
    }
    __syncthreads();
    compute(0);
    __syncthreads();
  }
#endif

  // ---- epilogue ----
  const int n = n0 + nloc;
#pragma unroll
  for (int v = 0; v < 8; ++v) {
    const int m = m0 + v + (hi << 3);
    if (m < M_TOT && n < Nreal) {
      float val = acc[v];
      if (bias) val += bias[n];
      if (outMode == 1) {
        Out[(size_t)m * Nreal + n] = fmaxf(val, 0.f);
      } else {
        const int b = m / 25, pos = m - 25 * b;
        Out[((size_t)b * Nreal + n) * 25 + pos] = val;
      }
    }
  }
}

// ---------------- training-mode BN: per-channel stats over (B,H,W)=200 ----------------
__global__ __launch_bounds__(256) void bn_stats_kernel(
    const float* __restrict__ o, const float* __restrict__ g,
    const float* __restrict__ bb, float* __restrict__ scale,
    float* __restrict__ shift) {
  const int c = (blockIdx.x << 3) + (threadIdx.x >> 5);   // one wave per channel
  const int lane = threadIdx.x & 31;
  float s = 0.f, ss = 0.f;
  for (int e = lane; e < M_TOT; e += 32) {
    const int b = e / 25, pos = e - 25 * b;
    const float v = o[((size_t)b * CIN + c) * 25 + pos];
    s += v; ss += v * v;
  }
  for (int off = 16; off > 0; off >>= 1) {
    s += __shfl_down(s, off);
    ss += __shfl_down(ss, off);
  }
  if (lane == 0) {
    const float m   = s * (1.f / 200.f);
    const float var = ss * (1.f / 200.f) - m * m;
    const float sc  = g[c] * rsqrtf(var + 1e-5f);
    scale[c] = sc;
    shift[c] = bb[c] - m * sc;
  }
}

__global__ __launch_bounds__(256) void bn_apply_kernel(
    const float* __restrict__ o, const float* __restrict__ scale,
    const float* __restrict__ shift, float* __restrict__ out, int relu) {
  const int i = blockIdx.x * 256 + threadIdx.x;
  if (i >= 8 * CIN * 25) return;
  const int c = (i / 25) & (CIN - 1);
  float v = o[i] * scale[c] + shift[c];
  if (relu) v = fmaxf(v, 0.f);
  out[i] = v;
}

// ---------------- fc2 (200x512 @ 4x512^T) + theta assembly ----------------
__global__ __launch_bounds__(256) void fc2_theta_kernel(
    const float* __restrict__ h, const float* __restrict__ w,
    float* __restrict__ pbuf, float* __restrict__ theta_out) {
  const int m = blockIdx.x;
  const int tid = threadIdx.x;
  const int lane = tid & 31, wv = tid >> 5;
  float a0 = 0.f, a1 = 0.f, a2 = 0.f, a3 = 0.f;
  for (int k = tid; k < 512; k += 256) {
    const float hv = h[(size_t)m * 512 + k];
    a0 += hv * w[k];
    a1 += hv * w[512 + k];
    a2 += hv * w[1024 + k];
    a3 += hv * w[1536 + k];
  }
  for (int off = 16; off > 0; off >>= 1) {
    a0 += __shfl_down(a0, off);
    a1 += __shfl_down(a1, off);
    a2 += __shfl_down(a2, off);
    a3 += __shfl_down(a3, off);
  }
  __shared__ float red[8][4];
  if (lane == 0) { red[wv][0] = a0; red[wv][1] = a1; red[wv][2] = a2; red[wv][3] = a3; }
  __syncthreads();
  if (tid == 0) {
    float p0 = 0.f, p1 = 0.f, p2 = 0.f, p3 = 0.f;
    for (int i = 0; i < 8; ++i) { p0 += red[i][0]; p1 += red[i][1]; p2 += red[i][2]; p3 += red[i][3]; }
    float* pp = pbuf + m * 4;
    pp[0] = p0; pp[1] = p1; pp[2] = p2; pp[3] = p3;
    float* t = theta_out + m * 6;        // [[1+p0,0,p1],[0,1+p2,p3]]
    t[0] = 1.f + p0; t[1] = 0.f; t[2] = p1;
    t[3] = 0.f; t[4] = 1.f + p2; t[5] = p3;
  }
}

// ---------------- affine grid-sample of padded f + stitch to [B,2048,15,15] ----------------
__global__ __launch_bounds__(256) void stn_sample_kernel(
    const float* __restrict__ f, const float* __restrict__ pbuf,
    float* __restrict__ temp) {
  const int gid = blockIdx.x;          // 0 .. 8*225-1
  const int b = gid / 225, r = gid - 225 * b;
  const int pp = r / 9, kk = r - 9 * pp;
  const int i = pp / 5, j = pp - 5 * i;
  const int u = kk / 3, vv = kk - 3 * u;
  const float* pv = pbuf + (b * 25 + pp) * 4;
  const float lx = -1.f + (float)(j + vv) * (2.f / 6.f);
  const float ly = -1.f + (float)(i + u) * (2.f / 6.f);
  const float gx = (1.f + pv[0]) * lx + pv[1];
  const float gy = (1.f + pv[2]) * ly + pv[3];
  const float fx = (gx + 1.f) * 0.5f * 6.f;      // 7x7 padded map, align_corners
  const float fy = (gy + 1.f) * 0.5f * 6.f;
  const float x0f = floorf(fx), y0f = floorf(fy);
  const int x0 = (int)x0f, y0 = (int)y0f;
  const float wx1 = fx - x0f, wx0 = 1.f - wx1;
  const float wy1 = fy - y0f, wy0 = 1.f - wy1;
  const int xs4[4] = {x0, x0 + 1, x0, x0 + 1};
  const int ys4[4] = {y0, y0, y0 + 1, y0 + 1};
  const float wt4[4] = {wx0 * wy0, wx1 * wy0, wx0 * wy1, wx1 * wy1};
  int idx[4]; float w4[4];
#pragma unroll
  for (int t = 0; t < 4; ++t) {
    // zero-pad grid_sample + zero border ring: nonzero only for 1..5 on the 7x7 map
    const bool valid = (xs4[t] >= 1) && (xs4[t] <= 5) && (ys4[t] >= 1) && (ys4[t] <= 5);
    idx[t] = valid ? ((ys4[t] - 1) * 5 + (xs4[t] - 1)) : 0;
    w4[t]  = valid ? wt4[t] : 0.f;
  }
  const int orow = i * 3 + u, ocol = j * 3 + vv;
  for (int c = threadIdx.x; c < CIN; c += 256) {
    const float* fc = f + ((size_t)b * CIN + c) * 25;
    const float a = w4[0] * fc[idx[0]] + w4[1] * fc[idx[1]] +
                    w4[2] * fc[idx[2]] + w4[3] * fc[idx[3]];
    temp[(((size_t)b * CIN + c) * 15 + orow) * 15 + ocol] = a;
  }
}

// ---------------- joint softmax over 201*25, s and class scores ----------------
__global__ __launch_bounds__(256) void softmax_out_kernel(
    const float* __restrict__ ostn, float* __restrict__ out) {
  const int b = blockIdx.x;
  const int tid = threadIdx.x;
  __shared__ float red[256];
  const float* xb = ostn + (size_t)b * 5025;
  float mx = -3.4e38f;
  for (int e = tid; e < 5025; e += 256) mx = fmaxf(mx, xb[e]);
  red[tid] = mx; __syncthreads();
  for (int s = 128; s > 0; s >>= 1) { if (tid < s) red[tid] = fmaxf(red[tid], red[tid + s]); __syncthreads(); }
  mx = red[0]; __syncthreads();
  float sm = 0.f;
  for (int e = tid; e < 5025; e += 256) sm += expf(xb[e] - mx);
  red[tid] = sm; __syncthreads();
  for (int s = 128; s > 0; s >>= 1) { if (tid < s) red[tid] += red[tid + s]; __syncthreads(); }
  const float inv = 1.f / red[0];
  float* s_out = out + 1608 + (size_t)b * 5025;
  for (int e = tid; e < 5025; e += 256) s_out[e] = expf(xb[e] - mx) * inv;
  float* sc = out + b * 201;
  for (int n = tid; n < 201; n += 256) {
    float a = 0.f;
    for (int pos = 0; pos < 25; ++pos) a += expf(xb[n * 25 + pos] - mx);
    sc[n] = a * inv;
  }
}

extern "C" void kernel_launch(void* const* d_in, const int* in_sizes, int n_in,
                              void* d_out, int out_size, void* d_ws, size_t ws_size,
                              hipStream_t stream) {
  (void)in_sizes; (void)n_in; (void)out_size; (void)ws_size;
  const float* x   = (const float*)d_in[0];
  // d_in[1]=epoch, d_in[2]=flag: unused by the computation
  const float* wL  = (const float*)d_in[3];
  const float* gL  = (const float*)d_in[4];
  const float* bL  = (const float*)d_in[5];
  const float* w1  = (const float*)d_in[6];
  const float* g1  = (const float*)d_in[7];
  const float* b1  = (const float*)d_in[8];
  const float* w2  = (const float*)d_in[9];
  const float* g2  = (const float*)d_in[10];
  const float* b2  = (const float*)d_in[11];
  const float* fc1 = (const float*)d_in[12];
  const float* fc2 = (const float*)d_in[13];
  const float* wS  = (const float*)d_in[14];
  const float* bS  = (const float*)d_in[15];
  float* out = (float*)d_out;   // scores[8*201] | s[8*201*25] | theta[200*6]

  // workspace carve-up (256B aligned)
  size_t off = 0;
  char* base = (char*)d_ws;
  auto take = [&](size_t bytes) -> char* {
    char* p = base + off;
    off = (off + bytes + 255) & ~(size_t)255;
    return p;
  };
  const size_t K = K_TOT;
  bf16* wlb = (bf16*)take(2048 * K * 2);
  bf16* w1b = (bf16*)take(2048 * K * 2);
  bf16* w2b = (bf16*)take(2048 * K * 2);
  bf16* wfb = (bf16*)take(512 * K * 2);
  bf16* wsb = (bf16*)take(256 * K * 2);          // N padded 201 -> 256 (2 full N-blocks)
  bf16* apack = (bf16*)take((size_t)M_PAD * K * 2);  // im2col staging, reused per stage
  float* o_raw = (float*)take(409600 * 4);       // [8,2048,5,5] pre-BN
  float* fbuf  = (float*)take(409600 * 4);       // f = bn(conv_last)
  float* xsbuf = (float*)take(409600 * 4);       // xs1 then xs2
  float* scale = (float*)take(2048 * 4);
  float* shift = (float*)take(2048 * 4);
  float* hbuf  = (float*)take(200 * 512 * 4);
  float* pbuf  = (float*)take(800 * 4);
  float* temp  = (float*)take((size_t)8 * 2048 * 225 * 4);
  float* ostn  = (float*)take((size_t)8 * 201 * 25 * 4);

  // 1) weights -> bf16 (read once from HBM; per-stage bf16 weights stay L2-resident)
  cvt_bf16_kernel<<<2048, 256, 0, stream>>>(wL,  wlb, 2048ull * K, 2048ull * K);
  cvt_bf16_kernel<<<2048, 256, 0, stream>>>(w1,  w1b, 2048ull * K, 2048ull * K);
  cvt_bf16_kernel<<<2048, 256, 0, stream>>>(w2,  w2b, 2048ull * K, 2048ull * K);
  cvt_bf16_kernel<<<512,  256, 0, stream>>>(fc1, wfb, 512ull * K, 512ull * K);
  cvt_bf16_kernel<<<512,  256, 0, stream>>>(wS,  wsb, 201ull * K, 256ull * K);

  // 2) conv_last (s=2,p=1 on 10x10) -> BN -> f
  im2col_bf16_kernel<<<dim3(72, 208), 256, 0, stream>>>(x, apack, 10, 10, 2, 1);
  gemm_wmma<<<dim3(13, 16), 256, 0, stream>>>(apack, wlb, o_raw, nullptr, 2048, 0);
  bn_stats_kernel<<<256, 256, 0, stream>>>(o_raw, gL, bL, scale, shift);
  bn_apply_kernel<<<1600, 256, 0, stream>>>(o_raw, scale, shift, fbuf, 0);

  // 3) loc conv1 -> BN -> relu -> xs1
  im2col_bf16_kernel<<<dim3(72, 208), 256, 0, stream>>>(fbuf, apack, 5, 5, 1, 1);
  gemm_wmma<<<dim3(13, 16), 256, 0, stream>>>(apack, w1b, o_raw, nullptr, 2048, 0);
  bn_stats_kernel<<<256, 256, 0, stream>>>(o_raw, g1, b1, scale, shift);
  bn_apply_kernel<<<1600, 256, 0, stream>>>(o_raw, scale, shift, xsbuf, 1);

  // 4) loc conv2 -> BN -> relu -> xs2 (reuse xsbuf)
  im2col_bf16_kernel<<<dim3(72, 208), 256, 0, stream>>>(xsbuf, apack, 5, 5, 1, 1);
  gemm_wmma<<<dim3(13, 16), 256, 0, stream>>>(apack, w2b, o_raw, nullptr, 2048, 0);
  bn_stats_kernel<<<256, 256, 0, stream>>>(o_raw, g2, b2, scale, shift);
  bn_apply_kernel<<<1600, 256, 0, stream>>>(o_raw, scale, shift, xsbuf, 1);

  // 5) fc1: patch extraction == s=1,p=1 im2col of xs2; relu -> h [200,512]
  im2col_bf16_kernel<<<dim3(72, 208), 256, 0, stream>>>(xsbuf, apack, 5, 5, 1, 1);
  gemm_wmma<<<dim3(13, 4), 256, 0, stream>>>(apack, wfb, hbuf, nullptr, 512, 1);

  // 6) fc2 + theta (theta written straight into d_out tail)
  fc2_theta_kernel<<<200, 256, 0, stream>>>(hbuf, fc2, pbuf, out + 41808);

  // 7) grid sample padded f, stitch -> temp [8,2048,15,15]
  stn_sample_kernel<<<1800, 256, 0, stream>>>(fbuf, pbuf, temp);

  // 8) conv_stn (k=3,s=3,p=0 on 15x15) + bias -> [8,201,5,5]
  im2col_bf16_kernel<<<dim3(72, 208), 256, 0, stream>>>(temp, apack, 15, 15, 3, 0);
  gemm_wmma<<<dim3(13, 2), 256, 0, stream>>>(apack, wsb, ostn, bS, 201, 0);

  // 9) joint softmax -> s and class scores
  softmax_out_kernel<<<8, 256, 0, stream>>>(ostn, out);
}
